// DeltaNet_31877247271467
// MI455X (gfx1250) — compile-verified, matrix operand order checked by reading, not run on hardware
//
#include <hip/hip_runtime.h>

typedef __bf16 bf16;
typedef __attribute__((ext_vector_type(2)))  float v2f;
typedef __attribute__((ext_vector_type(8)))  float v8f;
typedef __attribute__((ext_vector_type(8)))  bf16  v8bf;
typedef __attribute__((ext_vector_type(16))) bf16  v16bf;

#define BB   4
#define LL   4096
#define HS   1024
#define NH   4
#define DK   256
#define DV   256
#define DVH  128
#define CHK  32
#define MTOK (BB*LL)        /* 16384 */
#define MGATE (MTOK*NH)     /* 65536 */
#define GIN  1040
#define GH   1024

__device__ inline float siluf(float x) { return x / (1.f + expf(-x)); }
__device__ inline float geluf(float x) {
  float x3 = x * x * x;
  return 0.5f * x * (1.f + tanhf(0.79788456080286535f * (x + 0.044715f * x3)));
}

// ---------------- WMMA wrappers ----------------
__device__ inline v8f wmma_bf(v16bf a, v16bf b, v8f c) {
  return __builtin_amdgcn_wmma_f32_16x16x32_bf16(false, a, false, b, (short)0, c, false, false);
}
__device__ inline v8f wmma_f32(v2f a, v2f b, v8f c) {
  return __builtin_amdgcn_wmma_f32_16x16x4_f32(false, a, false, b, (short)0, c, false, false);
}

// f32 16x16x4 fragment helpers (LDS-resident matrices).
// A: a[e] = M[m0+lane%16][k0 + e + 2*(lane/16)]
__device__ inline v2f fragA32(const float* base, int ld, int m0, int k0, int lane) {
  int m  = m0 + (lane & 15);
  int kh = k0 + ((lane >> 4) << 1);
  v2f a; a.x = base[m * ld + kh]; a.y = base[m * ld + kh + 1];
  return a;
}
// B (matrix stored K x N row-major): b[e] = M[k0 + e + 2*half][n0+lane%16]
__device__ inline v2f fragB32(const float* base, int ld, int k0, int n0, int lane) {
  int n  = n0 + (lane & 15);
  int kh = k0 + ((lane >> 4) << 1);
  v2f b; b.x = base[kh * ld + n]; b.y = base[(kh + 1) * ld + n];
  return b;
}
__device__ inline void stC(float* base, int ld, int m0, int n0, int lane, v8f c) {
  int n = n0 + (lane & 15);
  int m = m0 + ((lane >> 4) << 3);
#pragma unroll
  for (int r = 0; r < 8; ++r) base[(m + r) * ld + n] = c[r];
}
__device__ inline v8f ldC(const float* base, int ld, int m0, int n0, int lane) {
  int n = n0 + (lane & 15);
  int m = m0 + ((lane >> 4) << 3);
  v8f c;
#pragma unroll
  for (int r = 0; r < 8; ++r) c[r] = base[(m + r) * ld + n];
  return c;
}
__device__ inline void subStoreC(float* base, int ld, int m0, int n0, int lane, v8f c) {
  int n = n0 + (lane & 15);
  int m = m0 + ((lane >> 4) << 3);
#pragma unroll
  for (int r = 0; r < 8; ++r) base[(m + r) * ld + n] -= c[r];
}

// bf16 16x16x32 fragment from LDS tile (row stride 40 bf16; two 16B runs per lane)
__device__ inline v16bf frag_bf(const bf16* tile, int row0, int lane) {
  int r = row0 + (lane & 15);
  int h = (lane >> 4) * 8;
  const bf16* p = tile + r * 40;
  v16bf out;
  *(v8bf*)&out        = *(const v8bf*)(p + h);
  *((v8bf*)&out + 1)  = *(const v8bf*)(p + 16 + h);
  return out;
}

// ---------------- elementwise f32 -> bf16 ----------------
__global__ __launch_bounds__(256) void f2bf_kernel(const float* __restrict__ in,
                                                   bf16* __restrict__ out, int n) {
  int i = blockIdx.x * 256 + threadIdx.x;
  if (i < n) out[i] = (bf16)in[i];
}

// ---------------- generic GEMM: C[M,N] = A[M,K] (bf16) @ B[N,K]^T (bf16) ----------------
template <bool OUT_BF>
__global__ __launch_bounds__(256)
void gemm_bf16_kernel(const bf16* __restrict__ A, const bf16* __restrict__ Bm,
                      void* __restrict__ Cout, int M, int N, int K) {
  __shared__ bf16 As[128 * 40];
  __shared__ bf16 Bs[64 * 40];
  int tid = threadIdx.x, lane = tid & 31, wv = tid >> 5;
  int bm = blockIdx.y * 128, bn = blockIdx.x * 64;
  int wm = (wv & 3) * 32, wn = (wv >> 2) * 32;
  v8f acc[2][2] = {{{}, {}}, {{}, {}}};
  for (int k0 = 0; k0 < K; k0 += 32) {
    int rr = tid >> 2, seg = tid & 3;
    bool valid = (k0 + seg * 8 + 8) <= K;
#pragma unroll
    for (int it = 0; it < 2; ++it) {
      int r = rr + it * 64;
      v8bf v = {};
      if (valid) v = *(const v8bf*)(A + (size_t)(bm + r) * K + k0 + seg * 8);
      *(v8bf*)(As + r * 40 + seg * 8) = v;
    }
    {
      v8bf v = {};
      if (valid) v = *(const v8bf*)(Bm + (size_t)(bn + rr) * K + k0 + seg * 8);
      *(v8bf*)(Bs + rr * 40 + seg * 8) = v;
    }
    __syncthreads();
    v16bf a0 = frag_bf(As, wm, lane),      a1 = frag_bf(As, wm + 16, lane);
    v16bf b0 = frag_bf(Bs, wn, lane),      b1 = frag_bf(Bs, wn + 16, lane);
    acc[0][0] = wmma_bf(a0, b0, acc[0][0]);
    acc[0][1] = wmma_bf(a0, b1, acc[0][1]);
    acc[1][0] = wmma_bf(a1, b0, acc[1][0]);
    acc[1][1] = wmma_bf(a1, b1, acc[1][1]);
    __syncthreads();
  }
#pragma unroll
  for (int i = 0; i < 2; ++i)
#pragma unroll
    for (int j = 0; j < 2; ++j) {
      int n = bn + wn + j * 16 + (lane & 15);
      int mBase = bm + wm + i * 16 + ((lane >> 4) << 3);
#pragma unroll
      for (int r = 0; r < 8; ++r) {
        size_t idx = (size_t)(mBase + r) * N + n;
        if (OUT_BF) ((bf16*)Cout)[idx] = (bf16)acc[i][j][r];
        else        ((float*)Cout)[idx] = acc[i][j][r];
      }
    }
}

// ---------------- gate GEMM: A = [hidden | stats] (virtual), fused GELU + W2 -> logits ----------------
__global__ __launch_bounds__(256)
void gate_gemm_kernel(const bf16* __restrict__ hbf, const bf16* __restrict__ statsbf,
                      const bf16* __restrict__ W1, const float* __restrict__ b1,
                      const float* __restrict__ W2, float* __restrict__ logits) {
  const int K = GIN, N = GH;
  __shared__ bf16 As[128 * 40];
  __shared__ bf16 Bs[64 * 40];
  int tid = threadIdx.x, lane = tid & 31, wv = tid >> 5;
  int bm = blockIdx.y * 128, bn = blockIdx.x * 64;
  int wm = (wv & 3) * 32, wn = (wv >> 2) * 32;
  v8f acc[2][2] = {{{}, {}}, {{}, {}}};
  for (int k0 = 0; k0 < K; k0 += 32) {
    int rr = tid >> 2, seg = tid & 3;
    int k = k0 + seg * 8;
#pragma unroll
    for (int it = 0; it < 2; ++it) {
      int r = rr + it * 64;
      int row = bm + r;
      v8bf v = {};
      if (k + 8 <= HS)       v = *(const v8bf*)(hbf + (size_t)(row >> 2) * HS + k);
      else if (k < GIN)      v = *(const v8bf*)(statsbf + (size_t)row * 16 + (k - HS));
      *(v8bf*)(As + r * 40 + seg * 8) = v;
    }
    {
      v8bf v = {};
      if (k + 8 <= K) v = *(const v8bf*)(W1 + (size_t)(bn + rr) * K + k);
      *(v8bf*)(Bs + rr * 40 + seg * 8) = v;
    }
    __syncthreads();
    v16bf a0 = frag_bf(As, wm, lane),      a1 = frag_bf(As, wm + 16, lane);
    v16bf b0 = frag_bf(Bs, wn, lane),      b1 = frag_bf(Bs, wn + 16, lane);
    acc[0][0] = wmma_bf(a0, b0, acc[0][0]);
    acc[0][1] = wmma_bf(a0, b1, acc[0][1]);
    acc[1][0] = wmma_bf(a1, b0, acc[1][0]);
    acc[1][1] = wmma_bf(a1, b1, acc[1][1]);
    __syncthreads();
  }
  // epilogue: h = gelu(acc + b1[n]); logits[m][g] += sum_n h * W2[g][n]
  int lane15 = lane & 15;
  int mh = (lane >> 4) << 3;
  float hv[2][2][8];
#pragma unroll
  for (int i = 0; i < 2; ++i)
#pragma unroll
    for (int j = 0; j < 2; ++j) {
      int n = bn + wn + j * 16 + lane15;
      float bb = b1[n];
#pragma unroll
      for (int r = 0; r < 8; ++r) hv[i][j][r] = geluf(acc[i][j][r] + bb);
    }
  int n0c = bn + wn + lane15;
#pragma unroll
  for (int i = 0; i < 2; ++i)
#pragma unroll
    for (int r = 0; r < 8; ++r) {
      int m = bm + wm + i * 16 + mh + r;
#pragma unroll
      for (int g = 0; g < 4; ++g) {
        float v = hv[i][0][r] * W2[g * GH + n0c] + hv[i][1][r] * W2[g * GH + n0c + 16];
        for (int mk = 1; mk < 16; mk <<= 1) v += __shfl_xor(v, mk, 32);
        if (lane15 == 0) atomicAdd(logits + (size_t)m * 4 + g, v);
      }
    }
}

// ---------------- beta = sigmoid(hidden @ Wb^T) ----------------
__global__ __launch_bounds__(256)
void beta_kernel(const float* __restrict__ hidden, const float* __restrict__ Wb,
                 float* __restrict__ beta) {
  int lane = threadIdx.x & 31, wv = threadIdx.x >> 5;
  int r = blockIdx.x * 8 + wv;            // (token, h)
  int token = r >> 2, h = r & 3;
  const float* hp = hidden + (size_t)token * HS;
  const float* wp = Wb + h * HS;
  float acc = 0.f;
#pragma unroll 8
  for (int j = 0; j < 32; ++j) { int d = lane + 32 * j; acc += hp[d] * wp[d]; }
  for (int m = 16; m >= 1; m >>= 1) acc += __shfl_xor(acc, m, 32);
  if (lane == 0) beta[r] = 1.f / (1.f + expf(-acc));
}

// ---------------- causal depthwise conv (K=4) + SiLU ----------------
template <bool OUT_BF>
__global__ __launch_bounds__(256)
void conv_silu_kernel(const bf16* __restrict__ pre, const float* __restrict__ w,
                      void* __restrict__ outp) {
  int idx = blockIdx.x * 256 + threadIdx.x;       // MTOK*HS
  int t = idx >> 10, c = idx & 1023;
  int l = t & (LL - 1);
  float acc = 0.f;
#pragma unroll
  for (int j = 0; j < 4; ++j) {
    int li = l - 3 + j;
    if (li >= 0) acc += w[c * 4 + j] * (float)pre[(size_t)(t - 3 + j) * 1024 + c];
  }
  float s = siluf(acc);
  if (OUT_BF) ((bf16*)outp)[idx] = (bf16)s;
  else        ((float*)outp)[idx] = s;
}

// ---------------- FIR depthwise causal conv (generic K<=64), f32 ----------------
__global__ __launch_bounds__(256)
void fir_kernel(const float* __restrict__ x, const float* __restrict__ filt,
                float* __restrict__ y, int Kf) {
  __shared__ float tile[127 * 64];
  __shared__ float fl[64 * 64];
  int tid = threadIdx.x;
  int cb = blockIdx.x & 15;
  int tb = blockIdx.x >> 4;
  int c0 = cb * 64, t0 = tb * 64;
  int l0 = t0 & (LL - 1);
  for (int i = tid; i < 64 * Kf; i += 256) {
    int c = i / Kf, j = i - c * Kf;
    fl[c * 64 + j] = filt[(size_t)(c0 + c) * Kf + j];
  }
  int nrows = Kf - 1 + 64;
  for (int i = tid; i < nrows * 64; i += 256) {
    int rr = i >> 6, c = i & 63;
    int l = l0 - (Kf - 1) + rr;
    tile[rr * 64 + c] = (l >= 0) ? x[(size_t)(t0 - (Kf - 1) + rr) * 1024 + c0 + c] : 0.f;
  }
  __syncthreads();
  int c = tid & 63;
  int rbase = (tid >> 6) * 16;
  for (int r = 0; r < 16; ++r) {
    float acc = 0.f;
    for (int j = 0; j < Kf; ++j) acc += fl[c * 64 + j] * tile[(rbase + r + j) * 64 + c];
    y[(size_t)(t0 + rbase + r) * 1024 + c0 + c] = acc;
  }
}

// ---------------- per-(b,l,h) stats of 4 sources -> bf16[16] ----------------
__global__ __launch_bounds__(256)
void stats_kernel(const float* __restrict__ fs, const float* __restrict__ flg,
                  const float* __restrict__ od, const float* __restrict__ vd,
                  bf16* __restrict__ statsbf) {
  int lane = threadIdx.x & 31, wv = threadIdx.x >> 5;
  int r = blockIdx.x * 8 + wv;
  int token = r >> 2, h = r & 3;
  size_t base = (size_t)token * 1024 + h * 256;
  const float* srcs[4] = {fs, flg, od, vd};
#pragma unroll
  for (int s = 0; s < 4; ++s) {
    const float* p = srcs[s] + base;
    float sm = 0.f, sa = 0.f, sq = 0.f;
#pragma unroll
    for (int j = 0; j < 8; ++j) {
      float v = p[lane + 32 * j];
      sm += v; sa += fabsf(v); sq += v * v;
    }
    for (int m = 16; m >= 1; m >>= 1) {
      sm += __shfl_xor(sm, m, 32);
      sa += __shfl_xor(sa, m, 32);
      sq += __shfl_xor(sq, m, 32);
    }
    if (lane == 0) {
      float mean = sm * (1.f / 256.f);
      statsbf[(size_t)r * 16 + s * 4 + 0] = (bf16)mean;
      statsbf[(size_t)r * 16 + s * 4 + 1] = (bf16)(sq * (1.f / 256.f) - mean * mean);
      statsbf[(size_t)r * 16 + s * 4 + 2] = (bf16)(sa * (1.f / 256.f));
      statsbf[(size_t)r * 16 + s * 4 + 3] = (bf16)sqrtf(sq);
    }
  }
}

// ---------------- gate softmax + mix + RMSNorm -> mixed (bf16) ----------------
__global__ __launch_bounds__(256)
void combine_kernel(const float* __restrict__ fs, const float* __restrict__ flg,
                    const float* __restrict__ od, const float* __restrict__ vd,
                    const float* __restrict__ logits, const float* __restrict__ b2,
                    const float* __restrict__ copy_bias, const float* __restrict__ log_temp,
                    const float* __restrict__ onw, bf16* __restrict__ mixed) {
  int lane = threadIdx.x & 31, wv = threadIdx.x >> 5;
  int r = blockIdx.x * 8 + wv;
  int token = r >> 2, h = r & 3;
  const float DEC = 1.f - 1.f / 3000.f;
  float lg[4];
#pragma unroll
  for (int g = 0; g < 4; ++g) lg[g] = logits[(size_t)r * 4 + g] + b2[g];
  lg[3] += copy_bias[h] * DEC;
  float it = 1.f / expf(log_temp[h]);
#pragma unroll
  for (int g = 0; g < 4; ++g) lg[g] *= it;
  float mx = fmaxf(fmaxf(lg[0], lg[1]), fmaxf(lg[2], lg[3]));
  float e[4], ssum = 0.f;
#pragma unroll
  for (int g = 0; g < 4; ++g) { e[g] = expf(lg[g] - mx); ssum += e[g]; }
  float eps = 0.08f * DEC;
  float wgt[4];
#pragma unroll
  for (int g = 0; g < 4; ++g) wgt[g] = (e[g] / ssum) * (1.f - 4.f * eps) + eps;
  size_t base = (size_t)token * 1024 + h * 256;
  float ov[8]; float sq = 0.f;
#pragma unroll
  for (int j = 0; j < 8; ++j) {
    int d = lane + 32 * j;
    float o = wgt[0] * fs[base + d] + wgt[1] * flg[base + d] +
              wgt[2] * od[base + d] + wgt[3] * vd[base + d];
    ov[j] = o; sq += o * o;
  }
  for (int m = 16; m >= 1; m >>= 1) sq += __shfl_xor(sq, m, 32);
  float scale = rsqrtf(sq * (1.f / 256.f) + 1e-5f);
#pragma unroll
  for (int j = 0; j < 8; ++j) {
    int d = lane + 32 * j;
    mixed[base + d] = (bf16)(ov[j] * scale * onw[d]);
  }
}

// ---------------- delta rule chunkwise scan (f32 WMMA, state in LDS) ----------------
// grid: 32 blocks = (b, h, dv-half). 256 threads = 8 waves.
__global__ __launch_bounds__(256)
void delta_kernel(const bf16* __restrict__ qg, const bf16* __restrict__ kg,
                  const float* __restrict__ vg, const float* __restrict__ beta_g,
                  float* __restrict__ o_d) {
  extern __shared__ float sm[];
  float* S   = sm;                 // 256 x 128
  float* Q   = S + DK * DVH;       // 32 x 256
  float* Kl  = Q + CHK * DK;       // 32 x 256
  float* W   = Kl + CHK * DK;      // 32 x 256 (kb -> w)
  float* U   = W + CHK * DK;       // 32 x 128 (vb -> u -> u_i)
  float* Tm  = U + CHK * DVH;      // 32 x 32
  float* C32 = Tm + CHK * CHK;     // 32 x 32 (M/A, then attn)
  float* bet = C32 + CHK * CHK;    // 32

  int tid = threadIdx.x, lane = tid & 31, wv = tid >> 5;
  int bidx = blockIdx.x;
  int b = bidx >> 3, h = (bidx >> 1) & 3, dhalf = bidx & 1;

  for (int i = tid; i < DK * DVH; i += 256) S[i] = 0.f;
  __syncthreads();

  for (int ch = 0; ch < LL / CHK; ++ch) {
    int tok0 = b * LL + ch * CHK;
    if (tid < CHK) bet[tid] = beta_g[(size_t)(tok0 + tid) * NH + h];
    // load q, k rows (bf16 -> f32) with row l2-norm
    {
      int row = tid >> 3, seg = tid & 7;
      const bf16* qr = qg + (size_t)(tok0 + row) * 1024 + h * DK + seg * 32;
      const bf16* kr = kg + (size_t)(tok0 + row) * 1024 + h * DK + seg * 32;
      float sq = 0.f, sk = 0.f;
#pragma unroll
      for (int j = 0; j < 32; ++j) {
        float qv = (float)qr[j]; Q[row * DK + seg * 32 + j] = qv; sq += qv * qv;
        float kv = (float)kr[j]; Kl[row * DK + seg * 32 + j] = kv; sk += kv * kv;
      }
      for (int m = 1; m < 8; m <<= 1) { sq += __shfl_xor(sq, m, 32); sk += __shfl_xor(sk, m, 32); }
      float rq = rsqrtf(sq + 1e-6f), rk = rsqrtf(sk + 1e-6f);
#pragma unroll
      for (int j = 0; j < 32; ++j) {
        Q[row * DK + seg * 32 + j]  *= rq;
        Kl[row * DK + seg * 32 + j] *= rk;
      }
    }
    __syncthreads();
    // W = k*beta ; U = v*beta (dv half)
    {
      int row = tid >> 3, seg = tid & 7;
      float bv = bet[row];
#pragma unroll
      for (int j = 0; j < 32; ++j)
        W[row * DK + seg * 32 + j] = Kl[row * DK + seg * 32 + j] * bv;
      if (seg < 4) {
        const float* vr = vg + (size_t)(tok0 + row) * 1024 + h * DV + dhalf * DVH + seg * 32;
#pragma unroll
        for (int j = 0; j < 32; ++j) U[row * DVH + seg * 32 + j] = vr[j] * bv;
      }
    }
    __syncthreads();
    // M = kb @ k^T  (32x32, K=256) -> C32
    if (wv < 4) {
      int m0 = (wv >> 1) * 16, n0 = (wv & 1) * 16;
      v8f acc = {};
      for (int kk = 0; kk < DK; kk += 4) {
        v2f a = fragA32(W, DK, m0, kk, lane);
        v2f bb = fragA32(Kl, DK, n0, kk, lane);   // B^T = k rows
        acc = wmma_f32(a, bb, acc);
      }
      stC(C32, CHK, m0, n0, lane, acc);
    }
    __syncthreads();
    // A = -strict_lower(M) (in place)
    for (int e = tid; e < CHK * CHK; e += 256) {
      int rr = e >> 5, jj = e & 31;
      float mv = C32[e];
      C32[e] = (jj < rr) ? -mv : 0.f;
    }
    __syncthreads();
    // T = (I - A)^{-1}: forward substitution by wave 0 (lane = column)
    if (wv == 0) {
      for (int rr = 0; rr < CHK; ++rr) {
        float acc = (rr == lane) ? 1.f : 0.f;
        for (int jj = 0; jj < rr; ++jj) acc += C32[rr * CHK + jj] * Tm[jj * CHK + lane];
        Tm[rr * CHK + lane] = acc;
      }
    }
    __syncthreads();
    // w = T @ kb (overwrite W, column-partitioned) ; u = T @ vb (overwrite U)
    {
      for (int nt = wv * 2; nt < wv * 2 + 2; ++nt) {
        int n0 = nt * 16;
        v8f a0 = {}, a1 = {};
        for (int kk = 0; kk < CHK; kk += 4) {
          v2f bb = fragB32(W, DK, kk, n0, lane);
          a0 = wmma_f32(fragA32(Tm, CHK, 0, kk, lane),  bb, a0);
          a1 = wmma_f32(fragA32(Tm, CHK, 16, kk, lane), bb, a1);
        }
        stC(W, DK, 0, n0, lane, a0);
        stC(W, DK, 16, n0, lane, a1);
      }
      int n0 = wv * 16;
      v8f a0 = {}, a1 = {};
      for (int kk = 0; kk < CHK; kk += 4) {
        v2f bb = fragB32(U, DVH, kk, n0, lane);
        a0 = wmma_f32(fragA32(Tm, CHK, 0, kk, lane),  bb, a0);
        a1 = wmma_f32(fragA32(Tm, CHK, 16, kk, lane), bb, a1);
      }
      stC(U, DVH, 0, n0, lane, a0);
      stC(U, DVH, 16, n0, lane, a1);
    }
    __syncthreads();
    // attn = tril(q @ k^T) -> C32
    if (wv < 4) {
      int m0 = (wv >> 1) * 16, n0 = (wv & 1) * 16;
      v8f acc = {};
      for (int kk = 0; kk < DK; kk += 4) {
        v2f a  = fragA32(Q, DK, m0, kk, lane);
        v2f bb = fragA32(Kl, DK, n0, kk, lane);
        acc = wmma_f32(a, bb, acc);
      }
      stC(C32, CHK, m0, n0, lane, acc);
    }
    __syncthreads();
    for (int e = tid; e < CHK * CHK; e += 256) {
      int rr = e >> 5, jj = e & 31;
      if (jj > rr) C32[e] = 0.f;
    }
    __syncthreads();
    // u_i = U - W @ S   (each wave owns columns n0 = wv*16 of S/U)
    {
      int n0 = wv * 16;
      v8f a0 = {}, a1 = {};
      for (int kk = 0; kk < DK; kk += 4) {
        v2f bb = fragB32(S, DVH, kk, n0, lane);
        a0 = wmma_f32(fragA32(W, DK, 0, kk, lane),  bb, a0);
        a1 = wmma_f32(fragA32(W, DK, 16, kk, lane), bb, a1);
      }
      subStoreC(U, DVH, 0, n0, lane, a0);
      subStoreC(U, DVH, 16, n0, lane, a1);
    }
    __syncthreads();
    // o = Q @ S + attn @ u_i ; store to global
    {
      int n0 = wv * 16;
      v8f a0 = {}, a1 = {};
      for (int kk = 0; kk < DK; kk += 4) {
        v2f bb = fragB32(S, DVH, kk, n0, lane);
        a0 = wmma_f32(fragA32(Q, DK, 0, kk, lane),  bb, a0);
        a1 = wmma_f32(fragA32(Q, DK, 16, kk, lane), bb, a1);
      }
      for (int kk = 0; kk < CHK; kk += 4) {
        v2f bb = fragB32(U, DVH, kk, n0, lane);
        a0 = wmma_f32(fragA32(C32, CHK, 0, kk, lane),  bb, a0);
        a1 = wmma_f32(fragA32(C32, CHK, 16, kk, lane), bb, a1);
      }
      int n = n0 + (lane & 15);
      int mh = ((lane >> 4) << 3);
      size_t col = (size_t)h * DV + dhalf * DVH + n;
#pragma unroll
      for (int r = 0; r < 8; ++r) {
        o_d[(size_t)(tok0 + mh + r) * 1024 + col]      = a0[r];
        o_d[(size_t)(tok0 + 16 + mh + r) * 1024 + col] = a1[r];
      }
    }
    __syncthreads();
    // S += k^T @ u_i (column slab per wave)
    {
      int n0 = wv * 16;
      for (int mt = 0; mt < 16; ++mt) {
        int m0 = mt * 16;
        v8f acc = ldC(S, DVH, m0, n0, lane);
        for (int kk = 0; kk < CHK; kk += 4) {
          v2f a  = fragB32(Kl, DK, kk, m0, lane);   // A = k^T
          v2f bb = fragB32(U, DVH, kk, n0, lane);
          acc = wmma_f32(a, bb, acc);
        }
        stC(S, DVH, m0, n0, lane, acc);
      }
    }
    __syncthreads();
  }
}

// =====================================================================
extern "C" void kernel_launch(void* const* d_in, const int* in_sizes, int n_in,
                              void* d_out, int out_size, void* d_ws, size_t ws_size,
                              hipStream_t stream) {
  (void)in_sizes; (void)n_in; (void)out_size; (void)ws_size;
  const float* hidden    = (const float*)d_in[0];
  const float* Wq        = (const float*)d_in[1];
  const float* Wk        = (const float*)d_in[2];
  const float* Wv        = (const float*)d_in[3];
  const float* Wb        = (const float*)d_in[4];
  const float* conv_q_w  = (const float*)d_in[5];
  const float* conv_k_w  = (const float*)d_in[6];
  const float* conv_v_w  = (const float*)d_in[7];
  const float* fir_short = (const float*)d_in[8];
  const float* fir_long  = (const float*)d_in[9];
  const float* gate_W1   = (const float*)d_in[10];
  const float* gate_b1   = (const float*)d_in[11];
  const float* gate_W2   = (const float*)d_in[12];
  const float* gate_b2   = (const float*)d_in[13];
  const float* copy_bias = (const float*)d_in[14];
  const float* log_temp  = (const float*)d_in[15];
  const float* o_norm_w  = (const float*)d_in[16];
  const float* Wo        = (const float*)d_in[17];

  char* ws = (char*)d_ws;
  size_t off = 0;
  auto alloc = [&](size_t bytes) {
    char* p = ws + off;
    off = (off + bytes + 255) & ~(size_t)255;
    return p;
  };
  bf16*  hbf    = (bf16*)alloc((size_t)MTOK * HS * 2);
  bf16*  wqb    = (bf16*)alloc((size_t)1024 * 1024 * 2);
  bf16*  wkb    = (bf16*)alloc((size_t)1024 * 1024 * 2);
  bf16*  wvb    = (bf16*)alloc((size_t)1024 * 1024 * 2);
  bf16*  w1b    = (bf16*)alloc((size_t)GH * GIN * 2);
  bf16*  wob    = (bf16*)alloc((size_t)1024 * 1024 * 2);
  bf16*  qpre   = (bf16*)alloc((size_t)MTOK * 1024 * 2);
  bf16*  kpre   = (bf16*)alloc((size_t)MTOK * 1024 * 2);
  bf16*  vpre   = (bf16*)alloc((size_t)MTOK * 1024 * 2);
  bf16*  qc     = (bf16*)alloc((size_t)MTOK * 1024 * 2);
  bf16*  kc     = (bf16*)alloc((size_t)MTOK * 1024 * 2);
  float* vc     = (float*)alloc((size_t)MTOK * 1024 * 4);
  float* betab  = (float*)alloc((size_t)MGATE * 4);
  float* od     = (float*)alloc((size_t)MTOK * 1024 * 4);
  float* firs   = (float*)alloc((size_t)MTOK * 1024 * 4);
  float* firl   = (float*)alloc((size_t)MTOK * 1024 * 4);
  bf16*  statsb = (bf16*)alloc((size_t)MGATE * 16 * 2);
  float* logitsb= (float*)alloc((size_t)MGATE * 4 * 4);
  bf16*  mixedb = (bf16*)alloc((size_t)MTOK * 1024 * 2);

  hipMemsetAsync(logitsb, 0, (size_t)MGATE * 4 * sizeof(float), stream);

  // f32 -> bf16 conversions
  int nh = MTOK * HS;
  f2bf_kernel<<<(nh + 255) / 256, 256, 0, stream>>>(hidden, hbf, nh);
  f2bf_kernel<<<(1024 * 1024 + 255) / 256, 256, 0, stream>>>(Wq, wqb, 1024 * 1024);
  f2bf_kernel<<<(1024 * 1024 + 255) / 256, 256, 0, stream>>>(Wk, wkb, 1024 * 1024);
  f2bf_kernel<<<(1024 * 1024 + 255) / 256, 256, 0, stream>>>(Wv, wvb, 1024 * 1024);
  f2bf_kernel<<<(GH * GIN + 255) / 256, 256, 0, stream>>>(gate_W1, w1b, GH * GIN);
  f2bf_kernel<<<(1024 * 1024 + 255) / 256, 256, 0, stream>>>(Wo, wob, 1024 * 1024);

  // q/k/v projections (WMMA)
  dim3 gproj(1024 / 64, MTOK / 128);
  gemm_bf16_kernel<true><<<gproj, 256, 0, stream>>>(hbf, wqb, qpre, MTOK, 1024, 1024);
  gemm_bf16_kernel<true><<<gproj, 256, 0, stream>>>(hbf, wkb, kpre, MTOK, 1024, 1024);
  gemm_bf16_kernel<true><<<gproj, 256, 0, stream>>>(hbf, wvb, vpre, MTOK, 1024, 1024);

  beta_kernel<<<MGATE / 8, 256, 0, stream>>>(hidden, Wb, betab);

  // causal conv + SiLU
  conv_silu_kernel<true><<<(MTOK * 1024) / 256, 256, 0, stream>>>(qpre, conv_q_w, qc);
  conv_silu_kernel<true><<<(MTOK * 1024) / 256, 256, 0, stream>>>(kpre, conv_k_w, kc);
  conv_silu_kernel<false><<<(MTOK * 1024) / 256, 256, 0, stream>>>(vpre, conv_v_w, vc);

  // delta-rule scan (f32 WMMA, 254KB dynamic LDS)
  const int DELTA_LDS = (DK * DVH + 3 * CHK * DK + CHK * DVH + 2 * CHK * CHK + CHK) * 4;
  hipFuncSetAttribute(reinterpret_cast<const void*>(delta_kernel),
                      hipFuncAttributeMaxDynamicSharedMemorySize, DELTA_LDS);
  delta_kernel<<<32, 256, DELTA_LDS, stream>>>(qc, kc, vc, betab, od);

  // FIR convs
  fir_kernel<<<(MTOK / 64) * 16, 256, 0, stream>>>(vc, fir_short, firs, 5);
  fir_kernel<<<(MTOK / 64) * 16, 256, 0, stream>>>(vc, fir_long, firl, 64);

  stats_kernel<<<MGATE / 8, 256, 0, stream>>>(firs, firl, od, vc, statsb);

  // gate MLP (WMMA + fused GELU + W2 reduction into logits)
  dim3 ggate(GH / 64, MGATE / 128);
  gate_gemm_kernel<<<ggate, 256, 0, stream>>>(hbf, statsb, w1b, gate_b1, gate_W2, logitsb);

  combine_kernel<<<MGATE / 8, 256, 0, stream>>>(firs, firl, od, vc, logitsb, gate_b2,
                                                copy_bias, log_temp, o_norm_w, mixedb);

  // output projection
  gemm_bf16_kernel<false><<<gproj, 256, 0, stream>>>(mixedb, wob, d_out, MTOK, 1024, 1024);
}
